// TriangleAttentionStartingNode_53300544143979
// MI455X (gfx1250) — compile-verified, matrix operand order checked by reading, not run on hardware
//
#include <hip/hip_runtime.h>
#include <hip/hip_bf16.h>

// TriangleAttentionStartingNode for MI455X (gfx1250), bf16 WMMA pipeline.
// B=1, N=256, c_z=128, H=4, c=32.

#define NN 256
#define CZ 128
#define NH 4
#define CH 32
#define HC 128

typedef __attribute__((ext_vector_type(16))) __bf16 v16bf;
typedef __attribute__((ext_vector_type(8)))  float  v8f;

union FragU { v16bf v; uint4 q[2]; };

// Load one 16-bit A/B WMMA fragment: this lane's 16 bf16 values are two
// contiguous 8-element runs at k = kbase and k = kbase+16 (ISA 7.12.2 layout:
// lanes 0-15 -> kbase 0 (K 0..7,16..23), lanes 16-31 -> kbase 8).
__device__ __forceinline__ v16bf load_frag(const __bf16* p, int kbase) {
  FragU f;
  f.q[0] = *reinterpret_cast<const uint4*>(p + kbase);
  f.q[1] = *reinterpret_cast<const uint4*>(p + kbase + 16);
  return f.v;
}

__device__ __forceinline__ v8f wmma_bf16(v16bf a, v16bf b, v8f c) {
  return __builtin_amdgcn_wmma_f32_16x16x32_bf16(
      /*neg_a=*/false, a, /*neg_b=*/false, b,
      /*c_mod=*/(short)0, c, /*reuse_a=*/false, /*reuse_b=*/false);
}

// Async copy 16 bytes global -> LDS (ASYNCcnt tracked, no VGPR round-trip).
// lds_off is the wave-relative LDS byte address (low 32 bits of a generic
// pointer into the shared aperture).
__device__ __forceinline__ void async_g2l_b128(unsigned lds_off, const void* gp) {
  asm volatile("global_load_async_to_lds_b128 %0, %1, off"
               :: "v"(lds_off), "v"(gp) : "memory");
}
__device__ __forceinline__ void wait_asynccnt0() {
  asm volatile("s_wait_asynccnt 0x0" ::: "memory");
}

// ---------------------------------------------------------------------------
// Kernel 0: transpose weights to (out,in) and cast to bf16 so B-fragments are
// two contiguous 16B loads per lane.
// ---------------------------------------------------------------------------
__global__ void __launch_bounds__(256) prep_weights(
    const float* __restrict__ Wqkv, const float* __restrict__ Wg,
    const float* __restrict__ Wo,
    __bf16* __restrict__ WqkvT, __bf16* __restrict__ WgT, __bf16* __restrict__ WoT)
{
  const int idx = blockIdx.x * 256 + threadIdx.x;
  const int n_qkv = 384 * CZ;   // 49152
  const int n_sq  = HC * CZ;    // 16384
  if (idx < n_qkv) {
    const int o = idx / CZ, k = idx % CZ;
    WqkvT[idx] = (__bf16)Wqkv[k * 384 + o];
  } else if (idx < n_qkv + n_sq) {
    const int t = idx - n_qkv;
    const int o = t / CZ, k = t % CZ;
    WgT[t] = (__bf16)Wg[k * HC + o];
  } else if (idx < n_qkv + 2 * n_sq) {
    const int t = idx - n_qkv - n_sq;
    const int o = t / CZ, k = t % CZ;
    WoT[t] = (__bf16)Wo[k * CZ + o];
  }
}

// ---------------------------------------------------------------------------
// Kernel 1: fused LayerNorm + qkv/gate/bias projections.
// Block = 256 threads = 8 waves; each wave owns 16 rows (M=16), 128 rows/block.
// ---------------------------------------------------------------------------
__global__ void __launch_bounds__(256) proj_kernel(
    const float* __restrict__ z, const float* __restrict__ lng,
    const float* __restrict__ lnb,
    const __bf16* __restrict__ WqkvT, const float* __restrict__ Wb,
    const __bf16* __restrict__ WgT,
    __bf16* __restrict__ qw, __bf16* __restrict__ kw, __bf16* __restrict__ vw,
    float* __restrict__ biasw, __bf16* __restrict__ gw)
{
  extern __shared__ char smem[];
  __bf16* zn = reinterpret_cast<__bf16*>(smem);   // [128][136] bf16 (272B rows)
  const int tid  = threadIdx.x;
  const int lane = tid & 31;
  const int wave = tid >> 5;
  const int rowBase = blockIdx.x * 128;

  // --- LayerNorm: one row per iteration per wave; 32 lanes x 4 floats ---
  for (int rr = 0; rr < 16; ++rr) {
    const int row = wave * 16 + rr;
    const float4 x =
        reinterpret_cast<const float4*>(z + (size_t)(rowBase + row) * CZ)[lane];
    float s  = x.x + x.y + x.z + x.w;
    float s2 = x.x * x.x + x.y * x.y + x.z * x.z + x.w * x.w;
    #pragma unroll
    for (int d = 16; d >= 1; d >>= 1) {
      s  += __shfl_xor(s,  d, 32);
      s2 += __shfl_xor(s2, d, 32);
    }
    const float mu  = s * (1.0f / CZ);
    const float var = s2 * (1.0f / CZ) - mu * mu;
    const float rs  = rsqrtf(var + 1e-5f);
    const float4 g4 = reinterpret_cast<const float4*>(lng)[lane];
    const float4 b4 = reinterpret_cast<const float4*>(lnb)[lane];
    __bf16* dst = zn + row * 136 + lane * 4;
    dst[0] = (__bf16)((x.x - mu) * rs * g4.x + b4.x);
    dst[1] = (__bf16)((x.y - mu) * rs * g4.y + b4.y);
    dst[2] = (__bf16)((x.z - mu) * rs * g4.z + b4.z);
    dst[3] = (__bf16)((x.w - mu) * rs * g4.w + b4.w);
  }
  __syncthreads();

  const int m  = lane & 15;
  const int kb = (lane < 16) ? 0 : 8;
  v16bf a[4];
  #pragma unroll
  for (int kt = 0; kt < 4; ++kt)
    a[kt] = load_frag(zn + (wave * 16 + m) * 136 + kt * 32, kb);

  // --- qkv: 384 cols = 24 tiles; load all K-step B frags, then WMMA chain ---
  for (int nt = 0; nt < 24; ++nt) {
    v16bf b[4];
    #pragma unroll
    for (int kt = 0; kt < 4; ++kt)
      b[kt] = load_frag(WqkvT + (nt * 16 + m) * CZ + kt * 32, kb);
    v8f acc = {};
    #pragma unroll
    for (int kt = 0; kt < 4; ++kt)
      acc = wmma_bf16(a[kt], b[kt], acc);
    const int col = nt * 16 + m;
    const int sec = col >> 7;          // 0=q 1=k 2=v
    const int cc  = col & 127;
    const int hh  = cc >> 5;
    const int ch  = cc & 31;
    __bf16* dstp = (sec == 0) ? qw : (sec == 1) ? kw : vw;
    #pragma unroll
    for (int r = 0; r < 8; ++r) {
      const int grow = rowBase + wave * 16 + ((lane < 16) ? r : r + 8);
      const int i = grow >> 8, j = grow & 255;
      dstp[(((size_t)hh * NN + i) * NN + j) * CH + ch] = (__bf16)acc[r];
    }
  }

  // --- gate: 128 cols = 8 tiles, sigmoid applied here ---
  for (int nt = 0; nt < 8; ++nt) {
    v16bf b[4];
    #pragma unroll
    for (int kt = 0; kt < 4; ++kt)
      b[kt] = load_frag(WgT + (nt * 16 + m) * CZ + kt * 32, kb);
    v8f acc = {};
    #pragma unroll
    for (int kt = 0; kt < 4; ++kt)
      acc = wmma_bf16(a[kt], b[kt], acc);
    const int col = nt * 16 + m;
    #pragma unroll
    for (int r = 0; r < 8; ++r) {
      const int grow = rowBase + wave * 16 + ((lane < 16) ? r : r + 8);
      const float sg = 1.0f / (1.0f + __expf(-acc[r]));
      gw[(size_t)grow * HC + col] = (__bf16)sg;
    }
  }

  // --- pair bias: only 4 output cols, VALU dot is fine ---
  for (int t = tid; t < 128 * 4; t += 256) {
    const int lrow = t >> 2, hh = t & 3;
    float s = 0.0f;
    for (int k2 = 0; k2 < CZ; ++k2)
      s += (float)zn[lrow * 136 + k2] * Wb[k2 * NH + hh];
    const int grow = rowBase + lrow;                 // (j,k) pair of z
    biasw[((size_t)hh * NN + (grow >> 8)) * NN + (grow & 255)] = s;
  }
}

// ---------------------------------------------------------------------------
// Kernel 2: attention for one (h, i, j-half). K rows + V rows pulled into LDS
// with GLOBAL_LOAD_ASYNC_TO_LDS_B128 (ASYNCcnt), V then transposed LDS->LDS.
// Scores kept in registers (16 x v8f per wave), softmax via 16-lane-group
// shuffles, attn round-tripped through LDS as bf16 A-fragments.
// ---------------------------------------------------------------------------
__global__ void __launch_bounds__(256) attn_kernel(
    const __bf16* __restrict__ qw, const __bf16* __restrict__ kw,
    const __bf16* __restrict__ vw, const float* __restrict__ biasw,
    const __bf16* __restrict__ gw, __bf16* __restrict__ ogw)
{
  extern __shared__ char smem[];
  __bf16* Kb = reinterpret_cast<__bf16*>(smem); // [256][40]  (80B rows)
  __bf16* Vt = Kb + 256 * 40;                   // [32][264]  (528B rows)
  __bf16* At = Vt + 32 * 264;                   // [8][16][264] per-wave attn
  __bf16* Vs = At;                              // V staging aliases At (pre-use)
  const int tid = threadIdx.x, lane = tid & 31, wave = tid >> 5;
  const int jhalf = blockIdx.x, i = blockIdx.y, h = blockIdx.z;
  const size_t blkBase = (((size_t)h * NN + i) * NN) * CH;

  // --- async-stage K rows and V rows (4 x b128 per thread each) ---
  {
    const unsigned koff = (unsigned)(size_t)(Kb + tid * 40);
    const unsigned voff = (unsigned)(size_t)(Vs + tid * 40);
    const __bf16* kg = kw + blkBase + (size_t)tid * CH;
    const __bf16* vg = vw + blkBase + (size_t)tid * CH;
    #pragma unroll
    for (int w = 0; w < 4; ++w) {
      async_g2l_b128(koff + w * 16, kg + w * 8);
      async_g2l_b128(voff + w * 16, vg + w * 8);
    }
    wait_asynccnt0();
  }
  __syncthreads();

  // --- transpose V (LDS->LDS): Vt[c][k] = Vs[k][c] ---
  #pragma unroll
  for (int c = 0; c < CH; ++c) Vt[c * 264 + tid] = Vs[tid * 40 + c];
  __syncthreads();

  const int j0 = jhalf * 128 + wave * 16;
  const int m = lane & 15, kb = (lane < 16) ? 0 : 8;
  const v16bf aq = load_frag(qw + blkBase + (size_t)(j0 + m) * CH, kb);

  // scores: 16 j-rows x 256 k, c=32 is exactly one bf16 WMMA K-step
  v8f acc[16];
  #pragma unroll
  for (int kt4 = 0; kt4 < 4; ++kt4) {
    v16bf b[4];
    #pragma unroll
    for (int t = 0; t < 4; ++t)
      b[t] = load_frag(Kb + ((kt4 * 4 + t) * 16 + m) * 40, kb);
    #pragma unroll
    for (int t = 0; t < 4; ++t) {
      v8f zr = {};
      acc[kt4 * 4 + t] = wmma_bf16(aq, b[t], zr);
    }
  }

  const float sc = 0.17677669529663687f;   // 1/sqrt(32)
  #pragma unroll
  for (int r = 0; r < 8; ++r) {
    const int j = j0 + ((lane < 16) ? r : r + 8);
    const float* bp = biasw + ((size_t)h * NN + j) * NN;
    float mx = -3.0e38f;
    #pragma unroll
    for (int kt = 0; kt < 16; ++kt) {
      acc[kt][r] = acc[kt][r] * sc + bp[kt * 16 + m];
      mx = fmaxf(mx, acc[kt][r]);
    }
    #pragma unroll
    for (int d = 8; d >= 1; d >>= 1) mx = fmaxf(mx, __shfl_xor(mx, d, 32));
    float sum = 0.0f;
    #pragma unroll
    for (int kt = 0; kt < 16; ++kt) {
      const float e = __expf(acc[kt][r] - mx);
      acc[kt][r] = e;
      sum += e;
    }
    #pragma unroll
    for (int d = 8; d >= 1; d >>= 1) sum += __shfl_xor(sum, d, 32);
    const float inv = 1.0f / sum;
    #pragma unroll
    for (int kt = 0; kt < 16; ++kt) acc[kt][r] *= inv;
  }

  // round-trip attn through this wave's private LDS strip as bf16 A-fragments
  __bf16* aw = At + wave * 16 * 264;
  #pragma unroll
  for (int kt = 0; kt < 16; ++kt) {
    #pragma unroll
    for (int r = 0; r < 8; ++r)
      aw[((lane < 16) ? r : r + 8) * 264 + kt * 16 + m] = (__bf16)acc[kt][r];
  }

  // O = attn @ V  (K=256 in 8 steps, 2 c-tiles)
  v8f oacc[2] = {};
  #pragma unroll
  for (int ks2 = 0; ks2 < 8; ++ks2) {
    const v16bf aa  = load_frag(aw + m * 264 + ks2 * 32, kb);
    v16bf bb[2];
    #pragma unroll
    for (int ct = 0; ct < 2; ++ct)
      bb[ct] = load_frag(Vt + (ct * 16 + m) * 264 + ks2 * 32, kb);
    #pragma unroll
    for (int ct = 0; ct < 2; ++ct)
      oacc[ct] = wmma_bf16(aa, bb[ct], oacc[ct]);
  }

  // gate * O -> bf16 rows laid out [i][j][H*C] for the output GEMM
  #pragma unroll
  for (int ct = 0; ct < 2; ++ct) {
    #pragma unroll
    for (int r = 0; r < 8; ++r) {
      const int j = j0 + ((lane < 16) ? r : r + 8);
      const size_t gi = (((size_t)i * NN + j) * HC) + h * CH + ct * 16 + m;
      ogw[gi] = (__bf16)((float)gw[gi] * oacc[ct][r]);
    }
  }
}

// ---------------------------------------------------------------------------
// Kernel 3: output projection (65536 x 128) @ (128 x 128) -> f32
// ---------------------------------------------------------------------------
__global__ void __launch_bounds__(256) out_kernel(
    const __bf16* __restrict__ ogw, const __bf16* __restrict__ WoT,
    float* __restrict__ out)
{
  const int tid = threadIdx.x, lane = tid & 31, wave = tid >> 5;
  const int rowBase = blockIdx.x * 128;
  const int m = lane & 15, kb = (lane < 16) ? 0 : 8;
  v16bf a[4];
  #pragma unroll
  for (int kt = 0; kt < 4; ++kt)
    a[kt] = load_frag(ogw + (size_t)(rowBase + wave * 16 + m) * HC + kt * 32, kb);
  for (int nt = 0; nt < 8; ++nt) {
    v16bf b[4];
    #pragma unroll
    for (int kt = 0; kt < 4; ++kt)
      b[kt] = load_frag(WoT + (nt * 16 + m) * CZ + kt * 32, kb);
    v8f acc = {};
    #pragma unroll
    for (int kt = 0; kt < 4; ++kt)
      acc = wmma_bf16(a[kt], b[kt], acc);
    #pragma unroll
    for (int r = 0; r < 8; ++r) {
      const int grow = rowBase + wave * 16 + ((lane < 16) ? r : r + 8);
      out[(size_t)grow * CZ + nt * 16 + m] = acc[r];
    }
  }
}

// ---------------------------------------------------------------------------
extern "C" void kernel_launch(void* const* d_in, const int* in_sizes, int n_in,
                              void* d_out, int out_size, void* d_ws, size_t ws_size,
                              hipStream_t stream) {
  (void)in_sizes; (void)n_in; (void)out_size; (void)ws_size;
  const float* z    = (const float*)d_in[0];
  const float* lng  = (const float*)d_in[1];
  const float* lnb  = (const float*)d_in[2];
  const float* Wqkv = (const float*)d_in[3];
  const float* Wb   = (const float*)d_in[4];
  const float* Wg   = (const float*)d_in[5];
  const float* Wo   = (const float*)d_in[6];
  float* out = (float*)d_out;

  char* ws = (char*)d_ws;
  size_t off = 0;
  auto carve = [&](size_t bytes) -> char* {
    char* p = ws + off;
    off += (bytes + 255) & ~(size_t)255;
    return p;
  };
  __bf16* WqkvT = (__bf16*)carve((size_t)384 * CZ * 2);
  __bf16* WgT   = (__bf16*)carve((size_t)HC * CZ * 2);
  __bf16* WoT   = (__bf16*)carve((size_t)HC * CZ * 2);
  __bf16* qwp   = (__bf16*)carve((size_t)NH * NN * NN * CH * 2);   // 16 MB
  __bf16* kwp   = (__bf16*)carve((size_t)NH * NN * NN * CH * 2);   // 16 MB
  __bf16* vwp   = (__bf16*)carve((size_t)NH * NN * NN * CH * 2);   // 16 MB
  float*  biasw = (float*) carve((size_t)NH * NN * NN * 4);        //  1 MB
  __bf16* gwp   = (__bf16*)carve((size_t)NN * NN * HC * 2);        // 16 MB
  __bf16* ogwp  = (__bf16*)carve((size_t)NN * NN * HC * 2);        // 16 MB

  const int proj_lds = 128 * 136 * 2;                              // 34816 B
  const int attn_lds = (256 * 40 + 32 * 264 + 8 * 16 * 264) * 2;   // 104960 B
  (void)hipFuncSetAttribute((const void*)proj_kernel,
      hipFuncAttributeMaxDynamicSharedMemorySize, proj_lds);
  (void)hipFuncSetAttribute((const void*)attn_kernel,
      hipFuncAttributeMaxDynamicSharedMemorySize, attn_lds);

  prep_weights<<<(384 * CZ + 2 * HC * CZ + 255) / 256, 256, 0, stream>>>(
      Wqkv, Wg, Wo, WqkvT, WgT, WoT);
  proj_kernel<<<512, 256, proj_lds, stream>>>(
      z, lng, lnb, WqkvT, Wb, WgT, qwp, kwp, vwp, biasw, gwp);
  attn_kernel<<<dim3(2, NN, NH), 256, attn_lds, stream>>>(
      qwp, kwp, vwp, biasw, gwp, ogwp);
  out_kernel<<<512, 256, 0, stream>>>(ogwp, WoT, out);
}
